// CrossGCN_84739704750763
// MI455X (gfx1250) — matrix operand order repaired and checked by reading, not compiled, not run on hardware
//
#include <hip/hip_runtime.h>
#include <hip/hip_bf16.h>

typedef _Float16 half_t;
typedef __attribute__((ext_vector_type(16))) half_t v16h;
typedef __attribute__((ext_vector_type(8)))  half_t v8h;
typedef __attribute__((ext_vector_type(8)))  float  v8f;

// ---------------------------------------------------------------------------
// Batched WMMA GEMM with TRANSPOSED B operand:
//   C[b](MxN,f32) = act( A[b](MxK,f16 row-major) * Bt[b](NxK,f16 row-major)^T
//                        + bias + res )
// K must be a multiple of 32 (buffers zero-padded). One wave computes a
// 16x(16*NT) tile. Software-pipelined: next K-step fragments are loaded before
// the WMMAs consuming the current ones, so load latency overlaps the matrix
// pipe instead of forcing s_wait_loadcnt 0 before every v_wmma.
// blockDim = (32,4): 4 waves cover 4 consecutive M-tiles.
// ---------------------------------------------------------------------------
template<int NT>
__global__ void gemm16t_k(const half_t* __restrict__ A, int lda, long sA,
                          const half_t* __restrict__ Bt, int ldbt, long sB,
                          float* __restrict__ C, int ldc, long sC,
                          int M, int N, int K,
                          const float* __restrict__ bias,
                          const float* __restrict__ res, long sRes,
                          int relu)
{
    const int lane  = threadIdx.x;                        // 0..31 (wave32)
    const int tileM = blockIdx.x * blockDim.y + threadIdx.y;
    const int m0    = tileM * 16;
    if (m0 >= M) return;                                   // wave-uniform exit
    const int n0 = blockIdx.y * (16 * NT);
    const int b  = blockIdx.z;

    const half_t* Ab = A  + (long)b * sA;
    const half_t* Bb = Bt + (long)b * sB;

    const int hsel = lane >> 4;                            // 0: lanes 0-15, 1: 16-31
    int mrow = m0 + (lane & 15);
    if (mrow >= M) mrow = M - 1;                           // clamp; stores guarded
    const int nlane = n0 + (lane & 15);

    v8f z = {};
    v8f acc[NT];
#pragma unroll
    for (int j = 0; j < NT; ++j) acc[j] = z;

    // A 16-bit 16x32 layout: lanes 0-15 hold K {0..7,16..23}, lanes 16-31 hold
    // K {8..15,24..31} -> two contiguous 8-half runs at +0 and +16.
    const half_t* pa0 = Ab + (long)mrow * lda + hsel * 8;
    // B via Bt (NxK), same K order as A: lanes 0-15 hold K {0..7,16..23},
    // lanes 16-31 hold K {8..15,24..31} -> two 8-half runs at +0 and +16.
    const half_t* pb0 = Bb + (long)nlane * ldbt + hsel * 8;

    union F { v16h v; v8h h[2]; };

    // prologue: fragments for k0 = 0
    F a;
    a.h[0] = *(const v8h*)(pa0);
    a.h[1] = *(const v8h*)(pa0 + 16);
    F bf[NT];
#pragma unroll
    for (int j = 0; j < NT; ++j) {
        const half_t* pb = pb0 + (long)j * 16 * ldbt;
        bf[j].h[0] = *(const v8h*)(pb);
        bf[j].h[1] = *(const v8h*)(pb + 16);
    }

#pragma unroll 2
    for (int k0 = 32; k0 < K; k0 += 32) {
        // issue next-iteration loads first (latency overlap with WMMAs below)
        F an;
        an.h[0] = *(const v8h*)(pa0 + k0);
        an.h[1] = *(const v8h*)(pa0 + k0 + 16);
        F bn[NT];
#pragma unroll
        for (int j = 0; j < NT; ++j) {
            const half_t* pb = pb0 + (long)j * 16 * ldbt + k0;
            bn[j].h[0] = *(const v8h*)(pb);
            bn[j].h[1] = *(const v8h*)(pb + 16);
        }
        // consume current fragments
#pragma unroll
        for (int j = 0; j < NT; ++j)
            acc[j] = __builtin_amdgcn_wmma_f32_16x16x32_f16(false, a.v, false, bf[j].v,
                                                            (short)0, acc[j],
                                                            false, false);
        a = an;
#pragma unroll
        for (int j = 0; j < NT; ++j) bf[j] = bn[j];
    }

    // epilogue: last K-step
#pragma unroll
    for (int j = 0; j < NT; ++j)
        acc[j] = __builtin_amdgcn_wmma_f32_16x16x32_f16(false, a.v, false, bf[j].v,
                                                        (short)0, acc[j],
                                                        false, false);

    float*       Cb = C + (long)b * sC;
    const float* Rb = res ? (res + (long)b * sRes) : nullptr;
#pragma unroll
    for (int j = 0; j < NT; ++j) {
        const int ncol = n0 + j * 16 + (lane & 15);
        const float bc = bias ? bias[ncol] : 0.f;
#pragma unroll
        for (int r = 0; r < 8; ++r) {
            int m = m0 + r + hsel * 8;                     // D: VGPR r -> M=r / r+8
            if (m < M) {
                float v = acc[j][r] + bc;
                if (Rb)   v += Rb[(long)m * ldc + ncol];
                if (relu) v = v > 0.f ? v : 0.f;
                Cb[(long)m * ldc + ncol] = v;
            }
        }
    }
}

// ---------------------------------------------------------------------------
// Elementwise f32 -> f16
// ---------------------------------------------------------------------------
__global__ void cvt_k(const float* __restrict__ s, half_t* __restrict__ d, long n)
{
    long i = (long)blockIdx.x * blockDim.x + threadIdx.x;
    if (i < n) d[i] = (half_t)s[i];
}

// Transposing convert for weights: src f32 (K,N) -> dst f16 (N,K)
__global__ void cvtT_k(const float* __restrict__ s, half_t* __restrict__ d,
                       int K, int N)
{
    long i = (long)blockIdx.x * blockDim.x + threadIdx.x;
    if (i >= (long)K * N) return;
    int k = (int)(i % K);
    int n = (int)(i / K);
    d[i] = (half_t)s[(long)k * N + n];                    // d[n*K+k]
}

// Transposing pad-convert: src f32 (B,M,C) -> dst f16 (B,C,Mp); zero cols >= M
__global__ void cvtpadT_k(const float* __restrict__ s, half_t* __restrict__ d,
                          int M, int Mp, int C, long total)
{
    long i = (long)blockIdx.x * blockDim.x + threadIdx.x;
    if (i >= total) return;
    int r = (int)(i % Mp);
    int c = (int)((i / Mp) % C);
    int b = (int)(i / ((long)Mp * C));
    d[i] = (r < M) ? (half_t)s[((long)b * M + r) * C + c] : (half_t)0.f;
}

// ---------------------------------------------------------------------------
// Cross-adjacency normalization
// denom1[b][v] = sum_o A[b][o][v] + eps   (column sums of A_OV)
// ---------------------------------------------------------------------------
__global__ void coldenom_k(const float* __restrict__ A, float* __restrict__ d1,
                           int NO, int NV, int total)
{
    int i = blockIdx.x * blockDim.x + threadIdx.x;
    if (i >= total) return;
    int b = i / NV, v = i % NV;
    const float* p = A + (long)b * NO * NV + v;
    float s = 0.f;
    for (int o = 0; o < NO; ++o) s += p[(long)o * NV];
    d1[i] = s + 1e-5f;
}

// denom2[b][o] = sum_v A[b][o][v]/denom1[b][v] + eps
__global__ void rowdenom_k(const float* __restrict__ A, const float* __restrict__ d1,
                           float* __restrict__ d2, int NO, int NV, int total)
{
    int i = blockIdx.x * blockDim.x + threadIdx.x;
    if (i >= total) return;
    int b = i / NO, o = i % NO;
    const float* p  = A  + ((long)b * NO + o) * NV;
    const float* dd = d1 + (long)b * NV;
    float s = 0.f;
    for (int v = 0; v < NV; ++v) s += p[v] / dd[v];
    d2[i] = s + 1e-5f;
}

// vo16[b][v][o] = A[b][o][v] / denom1[b][v]   (A operand; K-padded to NOp)
__global__ void vo_k(const float* __restrict__ A, const float* __restrict__ d1,
                     half_t* __restrict__ vo, int NO, int NV, int NOp, long total)
{
    long i = (long)blockIdx.x * blockDim.x + threadIdx.x;
    if (i >= total) return;
    int o = (int)(i % NOp);
    int v = (int)((i / NOp) % NV);
    int b = (int)(i / ((long)NOp * NV));
    half_t val = (half_t)0.f;
    if (o < NO)
        val = (half_t)(A[((long)b * NO + o) * NV + v] / d1[(long)b * NV + v]);
    vo[i] = val;
}

// ov16[b][o][v] = A[b][o][v] / denom1[b][v] / denom2[b][o]  (K-padded to NVp)
__global__ void ov_k(const float* __restrict__ A, const float* __restrict__ d1,
                     const float* __restrict__ d2, half_t* __restrict__ ov,
                     int NO, int NV, int NVp, long total)
{
    long i = (long)blockIdx.x * blockDim.x + threadIdx.x;
    if (i >= total) return;
    int v = (int)(i % NVp);
    int o = (int)((i / NVp) % NO);
    int b = (int)(i / ((long)NVp * NO));
    half_t val = (half_t)0.f;
    if (v < NV)
        val = (half_t)(A[((long)b * NO + o) * NV + v]
                       / d1[(long)b * NV + v] / d2[(long)b * NO + o]);
    ov[i] = val;
}

// ---------------------------------------------------------------------------
// Attention scores: s1[b][n] = h.a1, s2[b][n] = h.a2  (H=128)
// ---------------------------------------------------------------------------
__global__ void scores_k(const float* __restrict__ h, const float* __restrict__ a1,
                         const float* __restrict__ a2, float* __restrict__ s1,
                         float* __restrict__ s2, int total)
{
    int i = blockIdx.x * blockDim.x + threadIdx.x;
    if (i >= total) return;
    const float* hr = h + (long)i * 128;
    float x = 0.f, y = 0.f;
#pragma unroll 4
    for (int k = 0; k < 128; ++k) { float v = hr[k]; x += v * a1[k]; y += v * a2[k]; }
    s1[i] = x; s2[i] = y;
}

// ---------------------------------------------------------------------------
// Masked-softmax attention row -> f16 (K-padded). One block per (row, batch).
// Row staged in LDS (max 1504 floats << 320KB per WGP).
// ---------------------------------------------------------------------------
__global__ void att_row_k(const float* __restrict__ adj, const float* __restrict__ s1,
                          const float* __restrict__ s2, half_t* __restrict__ att,
                          int N, int Np)
{
    __shared__ float e[1504];
    __shared__ float red[256];
    const int b = blockIdx.y, i = blockIdx.x, tid = threadIdx.x;
    const float* arow = adj + ((long)b * N + i) * N;
    const float* s2b  = s2 + (long)b * N;
    const float  s1i  = s1[(long)b * N + i];

    for (int j = tid; j < N; j += 256) {
        float x = s1i + s2b[j];
        x = x > 0.f ? x : 0.2f * x;                      // leaky_relu(0.2)
        e[j] = (arow[j] > 0.f) ? x : -9e15f;
    }
    __syncthreads();

    float m = -3.4e38f;
    for (int j = tid; j < N; j += 256) m = fmaxf(m, e[j]);
    red[tid] = m; __syncthreads();
    for (int s = 128; s > 0; s >>= 1) {
        if (tid < s) red[tid] = fmaxf(red[tid], red[tid + s]);
        __syncthreads();
    }
    const float mx = red[0]; __syncthreads();

    float sum = 0.f;
    for (int j = tid; j < N; j += 256) sum += __expf(e[j] - mx);
    red[tid] = sum; __syncthreads();
    for (int s = 128; s > 0; s >>= 1) {
        if (tid < s) red[tid] += red[tid + s];
        __syncthreads();
    }
    const float inv = 1.f / red[0];

    half_t* orow = att + ((long)b * N + i) * Np;
    for (int j = tid; j < Np; j += 256)
        orow[j] = (j < N) ? (half_t)(__expf(e[j] - mx) * inv) : (half_t)0.f;
}

// ---------------------------------------------------------------------------
// Host-side orchestration
// ---------------------------------------------------------------------------
static void gemm_launch(const half_t* A, int lda, long sA,
                        const half_t* Bt, int ldbt, long sB,
                        float* C, int ldc, long sC,
                        int M, int N, int K, int batch,
                        const float* bias, const float* res, long sRes,
                        int relu, hipStream_t st)
{
    dim3 blk(32, 4, 1);
    if (N % 64 == 0) {
        dim3 grd((unsigned)((M + 63) / 64), (unsigned)(N / 64), (unsigned)batch);
        gemm16t_k<4><<<grd, blk, 0, st>>>(A, lda, sA, Bt, ldbt, sB, C, ldc, sC,
                                          M, N, K, bias, res, sRes, relu);
    } else {
        dim3 grd((unsigned)((M + 63) / 64), (unsigned)(N / 32), (unsigned)batch);
        gemm16t_k<2><<<grd, blk, 0, st>>>(A, lda, sA, Bt, ldbt, sB, C, ldc, sC,
                                          M, N, K, bias, res, sRes, relu);
    }
}

static void cvt_launch(const float* s, half_t* d, long n, hipStream_t st)
{
    cvt_k<<<dim3((unsigned)((n + 255) / 256)), dim3(256), 0, st>>>(s, d, n);
}

static void cvtT_launch(const float* s, half_t* d, int K, int N, hipStream_t st)
{
    long n = (long)K * N;
    cvtT_k<<<dim3((unsigned)((n + 255) / 256)), dim3(256), 0, st>>>(s, d, K, N);
}

static void cvtpadT_launch(const float* s, half_t* d, int M, int Mp, int C, int Bb,
                           hipStream_t st)
{
    long total = (long)Bb * C * Mp;
    cvtpadT_k<<<dim3((unsigned)((total + 255) / 256)), dim3(256), 0, st>>>(s, d, M, Mp, C, total);
}

struct Ctx {
    half_t *hvT, *hoT, *v1T, *o1T, *tv16, *to16, *zv16, *zo16;
    half_t *vis16, *obj16, *attv16, *atto16, *vo16, *ov16;
    float *fvA, *foA, *fvB, *foB, *fvC, *foC, *s1v, *s2v, *s1o, *s2o;
    const half_t *h_g2oW1t, *h_g2oW2t, *h_o2gW1t, *h_o2gW2t;
    const float *g2o_b1, *g2o_b2, *o2g_b1, *o2g_b2;
    const float *vis_adj, *obj_adj;
    hipStream_t st;
};

static void run_round(const Ctx& c, const half_t* visIn, int KV, const half_t* WvT,
                      const half_t* objIn, int KO, const half_t* WoT,
                      const float* ava, const float* avb,
                      const float* aoa, const float* aob)
{
    const int B = 8, NV = 500, NVp = 512, NO = 1500, NOp = 1504, H = 128;

    // h = x @ W   (B operand: W^T, shape HxK)
    gemm_launch(visIn, KV, (long)NV * KV, WvT, KV, 0, c.fvA, H, (long)NV * H,
                NV, H, KV, B, nullptr, nullptr, 0, 0, c.st);
    gemm_launch(objIn, KO, (long)NO * KO, WoT, KO, 0, c.foA, H, (long)NO * H,
                NO, H, KO, B, nullptr, nullptr, 0, 0, c.st);
    // transposed+padded f16 copies of h: (B, H, Np)
    cvtpadT_launch(c.fvA, c.hvT, NV, NVp, H, B, c.st);
    cvtpadT_launch(c.foA, c.hoT, NO, NOp, H, B, c.st);

    // attention scores
    scores_k<<<dim3((B * NV + 255) / 256), dim3(256), 0, c.st>>>(c.fvA, ava, avb, c.s1v, c.s2v, B * NV);
    scores_k<<<dim3((B * NO + 255) / 256), dim3(256), 0, c.st>>>(c.foA, aoa, aob, c.s1o, c.s2o, B * NO);

    // masked softmax rows -> f16 attention matrices (A operands)
    att_row_k<<<dim3(NV, B), dim3(256), 0, c.st>>>(c.vis_adj, c.s1v, c.s2v, c.attv16, NV, NVp);
    att_row_k<<<dim3(NO, B), dim3(256), 0, c.st>>>(c.obj_adj, c.s1o, c.s2o, c.atto16, NO, NOp);

    // v1 = relu(att @ h):  B operand = hT (H x Np)
    gemm_launch(c.attv16, NVp, (long)NV * NVp, c.hvT, NVp, (long)H * NVp,
                c.fvB, H, (long)NV * H, NV, H, NVp, B, nullptr, nullptr, 0, 1, c.st);
    gemm_launch(c.atto16, NOp, (long)NO * NOp, c.hoT, NOp, (long)H * NOp,
                c.foB, H, (long)NO * H, NO, H, NOp, B, nullptr, nullptr, 0, 1, c.st);
    cvtpadT_launch(c.fvB, c.v1T, NV, NVp, H, B, c.st);
    cvtpadT_launch(c.foB, c.o1T, NO, NOp, H, B, c.st);

    // cross aggregation: tv = vo @ o1 ; to = ov @ v1
    gemm_launch(c.vo16, NOp, (long)NV * NOp, c.o1T, NOp, (long)H * NOp,
                c.fvA, H, (long)NV * H, NV, H, NOp, B, nullptr, nullptr, 0, 0, c.st);
    gemm_launch(c.ov16, NVp, (long)NO * NVp, c.v1T, NVp, (long)H * NVp,
                c.foA, H, (long)NO * H, NO, H, NVp, B, nullptr, nullptr, 0, 0, c.st);
    cvt_launch(c.fvA, c.tv16, (long)8 * NV * H, c.st);
    cvt_launch(c.foA, c.to16, (long)8 * NO * H, c.st);

    // MLP layer 1 (+bias, relu): vis path uses o2g_*, obj path uses g2o_*
    gemm_launch(c.tv16, H, (long)NV * H, c.h_o2gW1t, H, 0, c.fvA, H, (long)NV * H,
                NV, H, H, B, c.o2g_b1, nullptr, 0, 1, c.st);
    gemm_launch(c.to16, H, (long)NO * H, c.h_g2oW1t, H, 0, c.foA, H, (long)NO * H,
                NO, H, H, B, c.g2o_b1, nullptr, 0, 1, c.st);
    cvt_launch(c.fvA, c.zv16, (long)8 * NV * H, c.st);
    cvt_launch(c.foA, c.zo16, (long)8 * NO * H, c.st);

    // MLP layer 2 (+bias) + residual v1/o1
    gemm_launch(c.zv16, H, (long)NV * H, c.h_o2gW2t, H, 0, c.fvC, H, (long)NV * H,
                NV, H, H, B, c.o2g_b2, c.fvB, (long)NV * H, 0, c.st);
    gemm_launch(c.zo16, H, (long)NO * H, c.h_g2oW2t, H, 0, c.foC, H, (long)NO * H,
                NO, H, H, B, c.g2o_b2, c.foB, (long)NO * H, 0, c.st);
    cvt_launch(c.fvC, c.vis16, (long)8 * NV * H, c.st);
    cvt_launch(c.foC, c.obj16, (long)8 * NO * H, c.st);
}

extern "C" void kernel_launch(void* const* d_in, const int* in_sizes, int n_in,
                              void* d_out, int out_size, void* d_ws, size_t ws_size,
                              hipStream_t stream)
{
    (void)in_sizes; (void)n_in; (void)out_size; (void)ws_size;
    const int B = 8, NV = 500, NVp = 512, NO = 1500, NOp = 1504;
    const int H = 128, DV = 512, DO = 32;

    const float* vis_memory = (const float*)d_in[0];
    const float* obj_memory = (const float*)d_in[1];
    const float* vis_adj    = (const float*)d_in[2];
    const float* obj_adj    = (const float*)d_in[3];
    const float* A_OV       = (const float*)d_in[4];
    const float* Wv1  = (const float*)d_in[5];
    const float* av1a = (const float*)d_in[6];
    const float* av1b = (const float*)d_in[7];
    const float* Wv2  = (const float*)d_in[8];
    const float* av2a = (const float*)d_in[9];
    const float* av2b = (const float*)d_in[10];
    const float* Wo1  = (const float*)d_in[11];
    const float* ao1a = (const float*)d_in[12];
    const float* ao1b = (const float*)d_in[13];
    const float* Wo2  = (const float*)d_in[14];
    const float* ao2a = (const float*)d_in[15];
    const float* ao2b = (const float*)d_in[16];
    const float* g2o_W1 = (const float*)d_in[17];
    const float* g2o_b1 = (const float*)d_in[18];
    const float* g2o_W2 = (const float*)d_in[19];
    const float* g2o_b2 = (const float*)d_in[20];
    const float* o2g_W1 = (const float*)d_in[21];
    const float* o2g_b1 = (const float*)d_in[22];
    const float* o2g_W2 = (const float*)d_in[23];
    const float* o2g_b2 = (const float*)d_in[24];
    const float* img_W  = (const float*)d_in[25];
    const float* img_b  = (const float*)d_in[26];
    const float* obj_W  = (const float*)d_in[27];
    const float* obj_b  = (const float*)d_in[28];

    // --- workspace arena (256B-aligned carve) ---
    char* base = (char*)d_ws;
    size_t cur = 0;
    auto alloc = [&](size_t bytes) -> void* {
        void* p = base + cur;
        cur += (bytes + 255) & ~(size_t)255;
        return p;
    };

    half_t* xv16   = (half_t*)alloc((size_t)B * NV * DV * 2);
    half_t* xo16   = (half_t*)alloc((size_t)B * NO * DO * 2);
    half_t* hWv1T  = (half_t*)alloc((size_t)DV * H * 2);
    half_t* hWv2T  = (half_t*)alloc((size_t)H * H * 2);
    half_t* hWo1T  = (half_t*)alloc((size_t)DO * H * 2);
    half_t* hWo2T  = (half_t*)alloc((size_t)H * H * 2);
    half_t* hG1T   = (half_t*)alloc((size_t)H * H * 2);    // g2o_W1^T
    half_t* hG2T   = (half_t*)alloc((size_t)H * H * 2);    // g2o_W2^T
    half_t* hO1T   = (half_t*)alloc((size_t)H * H * 2);    // o2g_W1^T
    half_t* hO2T   = (half_t*)alloc((size_t)H * H * 2);    // o2g_W2^T
    half_t* hImgT  = (half_t*)alloc((size_t)H * DV * 2);   // (DV x H)
    half_t* hObjT  = (half_t*)alloc((size_t)H * DO * 2);   // (DO x H)
    half_t* vo16   = (half_t*)alloc((size_t)B * NV * NOp * 2);
    half_t* ov16   = (half_t*)alloc((size_t)B * NO * NVp * 2);
    half_t* attv16 = (half_t*)alloc((size_t)B * NV * NVp * 2);
    half_t* atto16 = (half_t*)alloc((size_t)B * NO * NOp * 2);
    half_t* hvT    = (half_t*)alloc((size_t)B * H * NVp * 2);   // (B,H,NVp)
    half_t* hoT    = (half_t*)alloc((size_t)B * H * NOp * 2);
    half_t* v1T    = (half_t*)alloc((size_t)B * H * NVp * 2);
    half_t* o1T    = (half_t*)alloc((size_t)B * H * NOp * 2);
    half_t* tv16   = (half_t*)alloc((size_t)B * NV * H * 2);
    half_t* to16   = (half_t*)alloc((size_t)B * NO * H * 2);
    half_t* zv16   = (half_t*)alloc((size_t)B * NV * H * 2);
    half_t* zo16   = (half_t*)alloc((size_t)B * NO * H * 2);
    half_t* vis16  = (half_t*)alloc((size_t)B * NV * H * 2);
    half_t* obj16  = (half_t*)alloc((size_t)B * NO * H * 2);

    float* d1  = (float*)alloc((size_t)B * NV * 4);
    float* d2  = (float*)alloc((size_t)B * NO * 4);
    float* s1v = (float*)alloc((size_t)B * NV * 4);
    float* s2v = (float*)alloc((size_t)B * NV * 4);
    float* s1o = (float*)alloc((size_t)B * NO * 4);
    float* s2o = (float*)alloc((size_t)B * NO * 4);
    float* fvA = (float*)alloc((size_t)B * NV * H * 4);
    float* foA = (float*)alloc((size_t)B * NO * H * 4);
    float* fvB = (float*)alloc((size_t)B * NV * H * 4);
    float* foB = (float*)alloc((size_t)B * NO * H * 4);
    float* fvC = (float*)alloc((size_t)B * NV * H * 4);
    float* foC = (float*)alloc((size_t)B * NO * H * 4);

    // --- one-time conversions: inputs flat, weights transposed (N x K) ---
    cvt_launch(vis_memory, xv16, (long)B * NV * DV, stream);
    cvt_launch(obj_memory, xo16, (long)B * NO * DO, stream);
    cvtT_launch(Wv1, hWv1T, DV, H, stream);       // (DV,H) -> (H,DV)
    cvtT_launch(Wv2, hWv2T, H, H, stream);
    cvtT_launch(Wo1, hWo1T, DO, H, stream);       // (DO,H) -> (H,DO)
    cvtT_launch(Wo2, hWo2T, H, H, stream);
    cvtT_launch(g2o_W1, hG1T, H, H, stream);
    cvtT_launch(g2o_W2, hG2T, H, H, stream);
    cvtT_launch(o2g_W1, hO1T, H, H, stream);
    cvtT_launch(o2g_W2, hO2T, H, H, stream);
    cvtT_launch(img_W, hImgT, H, DV, stream);     // (H,DV) -> (DV,H)
    cvtT_launch(obj_W, hObjT, H, DO, stream);     // (H,DO) -> (DO,H)

    // --- cross adjacency normalization ---
    coldenom_k<<<dim3((B * NV + 255) / 256), dim3(256), 0, stream>>>(A_OV, d1, NO, NV, B * NV);
    rowdenom_k<<<dim3((B * NO + 255) / 256), dim3(256), 0, stream>>>(A_OV, d1, d2, NO, NV, B * NO);
    {
        long tot = (long)B * NV * NOp;
        vo_k<<<dim3((unsigned)((tot + 255) / 256)), dim3(256), 0, stream>>>(A_OV, d1, vo16, NO, NV, NOp, tot);
        tot = (long)B * NO * NVp;
        ov_k<<<dim3((unsigned)((tot + 255) / 256)), dim3(256), 0, stream>>>(A_OV, d1, d2, ov16, NO, NV, NVp, tot);
    }

    Ctx c;
    c.hvT = hvT; c.hoT = hoT; c.v1T = v1T; c.o1T = o1T;
    c.tv16 = tv16; c.to16 = to16; c.zv16 = zv16; c.zo16 = zo16;
    c.vis16 = vis16; c.obj16 = obj16; c.attv16 = attv16; c.atto16 = atto16;
    c.vo16 = vo16; c.ov16 = ov16;
    c.fvA = fvA; c.foA = foA; c.fvB = fvB; c.foB = foB; c.fvC = fvC; c.foC = foC;
    c.s1v = s1v; c.s2v = s2v; c.s1o = s1o; c.s2o = s2o;
    c.h_g2oW1t = hG1T; c.h_g2oW2t = hG2T; c.h_o2gW1t = hO1T; c.h_o2gW2t = hO2T;
    c.g2o_b1 = g2o_b1; c.g2o_b2 = g2o_b2; c.o2g_b1 = o2g_b1; c.o2g_b2 = o2g_b2;
    c.vis_adj = vis_adj; c.obj_adj = obj_adj; c.st = stream;

    // round 1 (raw features) and round 2 (vis16/obj16 carriers)
    run_round(c, xv16, DV, hWv1T, xo16, DO, hWo1T, av1a, av1b, ao1a, ao1b);
    run_round(c, vis16, H, hWv2T, obj16, H, hWo2T, av2a, av2b, ao2a, ao2b);

    // --- final projections straight into d_out ---
    float* out_vis = (float*)d_out;                       // (B, NV, DV)
    float* out_obj = out_vis + (size_t)B * NV * DV;       // (B, NO, DO)
    gemm_launch(vis16, H, (long)NV * H, hImgT, H, 0, out_vis, DV, (long)NV * DV,
                NV, DV, H, B, img_b, nullptr, 0, 0, stream);
    gemm_launch(obj16, H, (long)NO * H, hObjT, H, 0, out_obj, DO, (long)NO * DO,
                NO, DO, H, B, obj_b, nullptr, 0, 0, stream);
}
